// TransformerLayerController_42408507081001
// MI455X (gfx1250) — matrix operand order repaired and checked by reading, not compiled, not run on hardware
//
#include <hip/hip_runtime.h>

typedef __bf16 bf16;
typedef __attribute__((ext_vector_type(16))) __bf16 v16bf;
typedef __attribute__((ext_vector_type(8)))  float  v8f;

namespace {
constexpr int H = 32, S = 2048, D = 128;
constexpr int BLOCK_M = 128;  // 4 waves x 32 rows
constexpr int BLOCK_N = 32;   // keys per inner step
constexpr float SM_SCALE = 0.08838834764831845f; // 1/sqrt(128)
constexpr float NEG_INF = -1e30f;

constexpr size_t N_QKV   = (size_t)H * S * D;          // 8,388,608
constexpr size_t OFF_ATTN = 0;
constexpr size_t OFF_KQ   = OFF_ATTN + N_QKV;
constexpr size_t OFF_KSC  = OFF_KQ  + N_QKV;           // [H,1,D]
constexpr size_t OFF_KSV  = OFF_KSC + (size_t)H * D;   // [H,D,8]
constexpr size_t OFF_KSI  = OFF_KSV + (size_t)H * D * 8;
constexpr size_t OFF_VQ   = OFF_KSI + (size_t)H * D * 8;
constexpr size_t OFF_VSC  = OFF_VQ  + N_QKV;           // [H,S,1]
constexpr size_t OFF_VSV  = OFF_VSC + (size_t)H * S;   // [H,S,8]
constexpr size_t OFF_VSI  = OFF_VSV + (size_t)H * S * 8;
} // namespace

// ---- packed bf16 helpers (VOP3P) ------------------------------------------
__device__ __forceinline__ unsigned pack_bf16(float lo, float hi) {
    unsigned d;
    asm("v_cvt_pk_bf16_f32 %0, %1, %2" : "=v"(d) : "v"(lo), "v"(hi));
    return d;
}
__device__ __forceinline__ unsigned pk_max_bf16(unsigned a, unsigned b) {
    unsigned d;
    asm("v_pk_max_num_bf16 %0, %1, %2" : "=v"(d) : "v"(a), "v"(b));
    return d;
}
__device__ __forceinline__ float bf16_lo_f32(unsigned p) {
    return __uint_as_float(p << 16);
}
__device__ __forceinline__ float bf16_hi_f32(unsigned p) {
    return __uint_as_float(p & 0xffff0000u);
}

// ---------------------------------------------------------------------------
// Flash attention (causal, prefill) with bf16 WMMA, f32 accumulate.
// 4 waves/block, 32 query rows per wave (2 A-tiles), 32 keys per inner step.
// Row sums computed on the matrix pipe via a ones-column WMMA.
// ---------------------------------------------------------------------------
__global__ __launch_bounds__(128, 1)
void attn_fwd_kernel(const float* __restrict__ Q, const float* __restrict__ K,
                     const float* __restrict__ V, float* __restrict__ O)
{
    __shared__ bf16 k_lds[BLOCK_N][D];             // [key][dim]
    __shared__ bf16 v_lds[D][BLOCK_N + 8];         // [dim][key], padded
    __shared__ bf16 p_lds[4][2][16][BLOCK_N + 4];  // per-wave P staging

    const int tid = threadIdx.x;
    const int w   = tid >> 5;        // wave id 0..3
    const int ln  = tid & 31;        // lane
    const int g   = ln >> 4;         // half-wave group (0/1)
    const int m   = ln & 15;         // lane-in-group

    const int m0    = blockIdx.x * BLOCK_M;
    const int h     = blockIdx.y;
    const int qbase = m0 + 32 * w;   // this wave's first query row

    // ---- Q as 2 x 4 bf16 A-fragments (16x32 each, K chunks over D) ----
    // A k order used here (contiguous, same as B): lanes 0-15 hold K {0..15}, lanes 16-31 {16..31}
    v16bf qa[2][4];
#pragma unroll
    for (int qt = 0; qt < 2; ++qt) {
        const float* qrow = Q + ((size_t)(h * S + qbase + 16 * qt + m)) * D;
#pragma unroll
        for (int c = 0; c < 4; ++c) {
#pragma unroll
            for (int e = 0; e < 8; ++e)
                qa[qt][c][e] = (bf16)qrow[c * 32 + 16 * g + e];
#pragma unroll
            for (int e = 0; e < 8; ++e)
                qa[qt][c][8 + e] = (bf16)qrow[c * 32 + 16 * g + 8 + e];
        }
    }

    v16bf ones;
#pragma unroll
    for (int e = 0; e < 16; ++e) ones[e] = (bf16)1.0f;

    v8f oacc[2][8];
    v8f lacc[2];
    float mrow[2][8];
#pragma unroll
    for (int qt = 0; qt < 2; ++qt) {
        lacc[qt] = {};
#pragma unroll
        for (int n = 0; n < 8; ++n) oacc[qt][n] = {};
#pragma unroll
        for (int r = 0; r < 8; ++r) mrow[qt][r] = NEG_INF;
    }

    const float* Kg = K + (size_t)h * S * D;
    const float* Vg = V + (size_t)h * S * D;

    const int nblocks = (m0 + BLOCK_M) / BLOCK_N;   // causal
    for (int blk = 0; blk < nblocks; ++blk) {
        const int n0 = blk * BLOCK_N;

        __syncthreads();
        // cooperative f32->bf16 stage of 32x128 K and V tiles (coalesced float4)
#pragma unroll
        for (int i = 0; i < 8; ++i) {
            const int f4 = i * 128 + tid;       // 0..1023 float4 slots
            const int r  = f4 >> 5;             // key row 0..31
            const int c4 = f4 & 31;             // float4 column
            const float4 kv = ((const float4*)(Kg + (size_t)(n0 + r) * D))[c4];
            k_lds[r][c4 * 4 + 0] = (bf16)kv.x;
            k_lds[r][c4 * 4 + 1] = (bf16)kv.y;
            k_lds[r][c4 * 4 + 2] = (bf16)kv.z;
            k_lds[r][c4 * 4 + 3] = (bf16)kv.w;
            const float4 vv = ((const float4*)(Vg + (size_t)(n0 + r) * D))[c4];
            v_lds[c4 * 4 + 0][r] = (bf16)vv.x;
            v_lds[c4 * 4 + 1][r] = (bf16)vv.y;
            v_lds[c4 * 4 + 2][r] = (bf16)vv.z;
            v_lds[c4 * 4 + 3][r] = (bf16)vv.w;
        }
        __syncthreads();

        // prefetch next K/V tile into cache while we compute this one
        if (blk + 1 < nblocks) {
            const size_t pofs = (size_t)(n0 + BLOCK_N + (tid >> 2)) * D + (tid & 3) * 32;
            __builtin_prefetch(Kg + pofs, 0, 1);
            __builtin_prefetch(Vg + pofs, 0, 1);
        }

        if (n0 > qbase + 31) continue;   // wave fully above diagonal

        // ---- S = Q * K^T : 2 qtiles x 2 ktiles, K-dim 128 in 4 chunks ----
        v8f s[2][2];
        s[0][0] = {}; s[0][1] = {}; s[1][0] = {}; s[1][1] = {};
#pragma unroll
        for (int c = 0; c < 4; ++c) {
            v16bf bk0, bk1;
            const int kk0 = 32 * c + 16 * g;   // B layout: K = 16*g + e
#pragma unroll
            for (int e = 0; e < 16; ++e) bk0[e] = k_lds[m][kk0 + e];
#pragma unroll
            for (int e = 0; e < 16; ++e) bk1[e] = k_lds[16 + m][kk0 + e];
#pragma unroll
            for (int qt = 0; qt < 2; ++qt) {
                s[qt][0] = __builtin_amdgcn_wmma_f32_16x16x32_bf16(
                    false, qa[qt][c], false, bk0, (short)0, s[qt][0], false, false);
                s[qt][1] = __builtin_amdgcn_wmma_f32_16x16x32_bf16(
                    false, qa[qt][c], false, bk1, (short)0, s[qt][1], false, false);
            }
        }

        // ---- scale, causal mask, online softmax ----
        const int colbase = n0 + m;
        v16bf pa[2];
#pragma unroll
        for (int qt = 0; qt < 2; ++qt) {
#pragma unroll
            for (int r = 0; r < 8; ++r) {
                const int row = qbase + 16 * qt + r + 8 * g;   // C layout M = r+8g
                float e0 = s[qt][0][r] * SM_SCALE;
                float e1 = s[qt][1][r] * SM_SCALE;
                if (colbase > row)      e0 = NEG_INF;
                if (colbase + 16 > row) e1 = NEG_INF;
                s[qt][0][r] = e0;
                s[qt][1][r] = e1;
            }
            // packed-bf16 row-max reduction across the 16-lane group
            unsigned pk[4];
#pragma unroll
            for (int j = 0; j < 4; ++j)
                pk[j] = pack_bf16(fmaxf(s[qt][0][2 * j],     s[qt][1][2 * j]),
                                  fmaxf(s[qt][0][2 * j + 1], s[qt][1][2 * j + 1]));
#pragma unroll
            for (int msk = 8; msk; msk >>= 1) {
#pragma unroll
                for (int j = 0; j < 4; ++j)
                    pk[j] = pk_max_bf16(pk[j], (unsigned)__shfl_xor((int)pk[j], msk, 32));
            }
#pragma unroll
            for (int r = 0; r < 8; ++r) {
                const float mcur = (r & 1) ? bf16_hi_f32(pk[r >> 1]) : bf16_lo_f32(pk[r >> 1]);
                const float mnew  = fmaxf(mrow[qt][r], mcur);
                const float alpha = __expf(mrow[qt][r] - mnew);
                const float p0 = __expf(s[qt][0][r] - mnew);
                const float p1 = __expf(s[qt][1][r] - mnew);
                mrow[qt][r] = mnew;
                lacc[qt][r] = lacc[qt][r] * alpha;
#pragma unroll
                for (int n = 0; n < 8; ++n)
                    oacc[qt][n][r] = oacc[qt][n][r] * alpha;
                p_lds[w][qt][r + 8 * g][m]      = (bf16)p0;
                p_lds[w][qt][r + 8 * g][16 + m] = (bf16)p1;
            }
            // P as A-fragment (wave-private LDS, wave-ordered)
#pragma unroll
            for (int e = 0; e < 8; ++e) pa[qt][e]     = p_lds[w][qt][m][16 * g + e];
#pragma unroll
            for (int e = 0; e < 8; ++e) pa[qt][8 + e] = p_lds[w][qt][m][16 * g + 8 + e];
            // row sums on the matrix pipe: l += P * ones
            lacc[qt] = __builtin_amdgcn_wmma_f32_16x16x32_bf16(
                false, pa[qt], false, ones, (short)0, lacc[qt], false, false);
        }

        // ---- O += P * V : V B-fragments shared by both query tiles ----
#pragma unroll
        for (int nt = 0; nt < 8; ++nt) {
            v16bf bv;
            const int k0 = 16 * g;                 // B layout: K(=key) = 16*g + e
#pragma unroll
            for (int e = 0; e < 16; ++e) bv[e] = v_lds[16 * nt + m][k0 + e];
            oacc[0][nt] = __builtin_amdgcn_wmma_f32_16x16x32_bf16(
                false, pa[0], false, bv, (short)0, oacc[0][nt], false, false);
            oacc[1][nt] = __builtin_amdgcn_wmma_f32_16x16x32_bf16(
                false, pa[1], false, bv, (short)0, oacc[1][nt], false, false);
        }
    }

    // ---- normalize & store (C layout: row = r + 8*g, col = 16*nt + m) ----
#pragma unroll
    for (int qt = 0; qt < 2; ++qt) {
        float* obase = O + ((size_t)(h * S + qbase + 16 * qt)) * D;
#pragma unroll
        for (int r = 0; r < 8; ++r) {
            const int row = r + 8 * g;
            const float inv = 1.0f / lacc[qt][r];
#pragma unroll
            for (int nt = 0; nt < 8; ++nt)
                obase[(size_t)row * D + 16 * nt + m] = oacc[qt][nt][r] * inv;
        }
    }
}

// ---------------------------------------------------------------------------
// K quantization: per (h,d) channel over 2048 tokens. One wave per channel.
// ---------------------------------------------------------------------------
__global__ __launch_bounds__(128)
void k_quant_kernel(const float* __restrict__ K, float* __restrict__ kq,
                    float* __restrict__ ksc, float* __restrict__ ksv,
                    float* __restrict__ ksi)
{
    __shared__ float ch[4][S];                 // 32 KB
    const int w = threadIdx.x >> 5, ln = threadIdx.x & 31;
    const int chan = blockIdx.x * 4 + w;       // h*128 + d
    const int h = chan >> 7, d = chan & 127;
    const float* base = K + (size_t)h * S * D + d;

    for (int j = 0; j < S / 32; ++j) {
        const int s = ln + 32 * j;
        ch[w][s] = (s < 4) ? 0.0f : base[(size_t)s * D];   // zero sink tokens
    }

    float spv[8]; int spi[8];
    for (int it = 0; it < 8; ++it) {
        float bm = -1.0f; int bi = 0;
        for (int j = 0; j < S / 32; ++j) {
            const int s = ln + 32 * j;
            const float a = fabsf(ch[w][s]);
            if (a > bm) { bm = a; bi = s; }
        }
#pragma unroll
        for (int msk = 16; msk; msk >>= 1) {
            const float om = __shfl_xor(bm, msk, 32);
            const int   oi = __shfl_xor(bi, msk, 32);
            if (om > bm || (om == bm && oi < bi)) { bm = om; bi = oi; }
        }
        spv[it] = ch[w][bi];
        spi[it] = bi;
        if ((bi & 31) == ln) ch[w][bi] = 0.0f;   // zero winner (wave-ordered LDS)
    }

    float am = 0.0f;
    for (int j = 0; j < S / 32; ++j)
        am = fmaxf(am, fabsf(ch[w][ln + 32 * j]));
#pragma unroll
    for (int msk = 16; msk; msk >>= 1)
        am = fmaxf(am, __shfl_xor(am, msk, 32));

    const float scale = am * (1.0f / 127.0f) + 1e-8f;
    const float inv = 1.0f / scale;
    if (ln == 0) {
        ksc[chan] = scale;
#pragma unroll
        for (int i = 0; i < 8; ++i) {
            ksv[(size_t)chan * 8 + i] = spv[i];
            ksi[(size_t)chan * 8 + i] = (float)spi[i];
        }
    }
    for (int j = 0; j < S / 32; ++j) {
        const int s = ln + 32 * j;
        const float q = fminf(127.0f, fmaxf(-127.0f, rintf(ch[w][s] * inv)));
        kq[((size_t)h * S + s) * D + d] = q;
    }
}

// ---------------------------------------------------------------------------
// V quantization: per (h,s) token over 128 channels. One wave per token row.
// ---------------------------------------------------------------------------
__global__ __launch_bounds__(128)
void v_quant_kernel(const float* __restrict__ V, float* __restrict__ vq,
                    float* __restrict__ vsc, float* __restrict__ vsv,
                    float* __restrict__ vsi)
{
    const int w = threadIdx.x >> 5, ln = threadIdx.x & 31;
    const int row = blockIdx.x * 4 + w;        // h*S + s
    const int s = row & (S - 1);

    float4 vv = ((const float4*)(V + (size_t)row * D))[ln];
    float v0 = vv.x, v1 = vv.y, v2 = vv.z, v3 = vv.w;
    if (s < 4) { v0 = v1 = v2 = v3 = 0.0f; }

    float spv[8]; int spi[8];
#pragma unroll
    for (int it = 0; it < 8; ++it) {
        float bm = fabsf(v0); int bi = 4 * ln;
        if (fabsf(v1) > bm) { bm = fabsf(v1); bi = 4 * ln + 1; }
        if (fabsf(v2) > bm) { bm = fabsf(v2); bi = 4 * ln + 2; }
        if (fabsf(v3) > bm) { bm = fabsf(v3); bi = 4 * ln + 3; }
#pragma unroll
        for (int msk = 16; msk; msk >>= 1) {
            const float om = __shfl_xor(bm, msk, 32);
            const int   oi = __shfl_xor(bi, msk, 32);
            if (om > bm || (om == bm && oi < bi)) { bm = om; bi = oi; }
        }
        const int owner = bi >> 2, sub = bi & 3;
        const float cand = (sub == 0) ? v0 : (sub == 1) ? v1 : (sub == 2) ? v2 : v3;
        spv[it] = __shfl(cand, owner, 32);
        spi[it] = bi;
        if (ln == owner) {
            if (sub == 0) v0 = 0.0f; else if (sub == 1) v1 = 0.0f;
            else if (sub == 2) v2 = 0.0f; else v3 = 0.0f;
        }
    }

    float am = fmaxf(fmaxf(fabsf(v0), fabsf(v1)), fmaxf(fabsf(v2), fabsf(v3)));
#pragma unroll
    for (int msk = 16; msk; msk >>= 1)
        am = fmaxf(am, __shfl_xor(am, msk, 32));

    const float scale = am * (1.0f / 127.0f) + 1e-8f;
    const float inv = 1.0f / scale;
    if (ln == 0) {
        vsc[row] = scale;
#pragma unroll
        for (int i = 0; i < 8; ++i) {
            vsv[(size_t)row * 8 + i] = spv[i];
            vsi[(size_t)row * 8 + i] = (float)spi[i];
        }
    }
    float4 q;
    q.x = fminf(127.0f, fmaxf(-127.0f, rintf(v0 * inv)));
    q.y = fminf(127.0f, fmaxf(-127.0f, rintf(v1 * inv)));
    q.z = fminf(127.0f, fmaxf(-127.0f, rintf(v2 * inv)));
    q.w = fminf(127.0f, fmaxf(-127.0f, rintf(v3 * inv)));
    ((float4*)(vq + (size_t)row * D))[ln] = q;
}

// ---------------------------------------------------------------------------
extern "C" void kernel_launch(void* const* d_in, const int* in_sizes, int n_in,
                              void* d_out, int out_size, void* d_ws, size_t ws_size,
                              hipStream_t stream)
{
    (void)in_sizes; (void)n_in; (void)out_size; (void)d_ws; (void)ws_size;
    const float* Q = (const float*)d_in[0];
    const float* K = (const float*)d_in[1];
    const float* V = (const float*)d_in[2];
    float* out = (float*)d_out;

    attn_fwd_kernel<<<dim3(S / BLOCK_M, H), 128, 0, stream>>>(Q, K, V, out + OFF_ATTN);

    k_quant_kernel<<<(H * D) / 4, 128, 0, stream>>>(K, out + OFF_KQ, out + OFF_KSC,
                                                    out + OFF_KSV, out + OFF_KSI);

    v_quant_kernel<<<(H * S) / 4, 128, 0, stream>>>(V, out + OFF_VQ, out + OFF_VSC,
                                                    out + OFF_VSV, out + OFF_VSI);
}
